// DiffusionLayer_82480551952522
// MI455X (gfx1250) — matrix operand order repaired and checked by reading, not compiled, not run on hardware
//
#include <hip/hip_runtime.h>

#define NROW 8192
#define DDIM 128
#define TAUF 0.1f
#define HSTR 136   // LDS row stride (halves) for MLP hidden relayout
#define PSTR 72    // LDS row stride (halves) for P relayout (>=64, mult of 8)
#define FRAGH 24   // halves per lane per staged fragment (48B: bank-optimal, 16B-aligned)
#define FRAGSZ (32 * FRAGH)        // halves per staged fragment (32 lanes)
#define STAGEH (16 * FRAGSZ)       // halves per stage buffer (16 frags)

typedef __attribute__((ext_vector_type(16))) _Float16 v16h;
typedef __attribute__((ext_vector_type(8)))  _Float16 v8h;
typedef __attribute__((ext_vector_type(4)))  _Float16 v4h;
typedef __attribute__((ext_vector_type(8)))  float    v8f;

// D = A(16x32 f16) * B(32x16 f16) + C(16x16 f32)
__device__ __forceinline__ v8f wmma16(v16h a, v16h b, v8f c) {
  return __builtin_amdgcn_wmma_f32_16x16x32_f16(false, a, false, b, (short)0, c,
                                                false, false);
}

// Low 32 bits of a flat shared-memory pointer == hardware LDS byte address.
__device__ __forceinline__ unsigned lds_low32(const void* p) {
  return (unsigned)(unsigned long long)p;
}

// A-layout fragment (16x32 f16): lane m=lane&15 holds row m.
__device__ __forceinline__ v16h load_afrag(const _Float16* __restrict__ src,
                                           int row0, int k0, int ld, int lane) {
  int m = lane & 15, hh = lane >> 4;
  const _Float16* p = src + (size_t)(row0 + m) * ld + k0;
  v8h x0 = *(const v8h*)(p + hh * 16);
  v8h x1 = *(const v8h*)(p + hh * 16 + 8);
  v16h r;
#pragma unroll
  for (int i = 0; i < 8; ++i) { r[i] = x0[i]; r[i + 8] = x1[i]; }
  return r;
}

// B-layout fragment (32x16 f16) from row-major transposed operand.
__device__ __forceinline__ v16h load_bfrag(const _Float16* __restrict__ src,
                                           int col0, int colstep, int k0,
                                           int ld, int lane) {
  int n = lane & 15, hh = lane >> 4;
  const _Float16* p = src + (size_t)(col0 + n * colstep) * ld + k0 + hh * 16;
  v8h x0 = *(const v8h*)(p);
  v8h x1 = *(const v8h*)(p + 8);
  v16h r;
#pragma unroll
  for (int i = 0; i < 8; ++i) { r[i] = x0[i]; r[i + 8] = x1[i]; }
  return r;
}

// B fragment from the swizzled LDS stage: lane's 16 halves are contiguous.
__device__ __forceinline__ v16h load_stage_frag(const _Float16* base, int lane) {
  const _Float16* p = base + lane * FRAGH;
  v8h x0 = *(const v8h*)(p);
  v8h x1 = *(const v8h*)(p + 8);
  v16h r;
#pragma unroll
  for (int i = 0; i < 8; ++i) { r[i] = x0[i]; r[i + 8] = x1[i]; }
  return r;
}

// Async-stage one 64-key tile of tH into LDS in fragment-swizzled layout.
// Frag f=(k<<2)|t holds B(32x16) for sim tile t, k-chunk k; each lane owns the
// 32 bytes it will later consume. 8 async b128 transfers per wave.
__device__ __forceinline__ void stage_keyblock(const _Float16* __restrict__ tH,
                                               _Float16* stageBuf, int kb,
                                               int wave, int lane) {
  int n = lane & 15, hh = lane >> 4;
#pragma unroll
  for (int q = 0; q < 4; ++q) {
    int f = wave * 4 + q;
    int t = f & 3, k = f >> 2;
    unsigned long long ga = (unsigned long long)(
        tH + (size_t)(kb + 4 * n + t) * DDIM + k * 32 + hh * 16);
    unsigned la = lds_low32(stageBuf + f * FRAGSZ + lane * FRAGH);
    asm volatile("global_load_async_to_lds_b128 %0, %1, off"
                 :: "v"(la), "v"(ga) : "memory");
    asm volatile("global_load_async_to_lds_b128 %0, %1, off"
                 :: "v"(la + 16u), "v"(ga + 16ull) : "memory");
  }
}

// ---------------- prep: build Z^T (f16) and transposed f16 weights ----------
__global__ void diff_prep_kernel(const float* __restrict__ Z,
                                 const float* __restrict__ W1,
                                 const float* __restrict__ W2,
                                 _Float16* __restrict__ ZT,
                                 _Float16* __restrict__ W1T,
                                 _Float16* __restrict__ W2T) {
  int i = blockIdx.x * blockDim.x + threadIdx.x;
  if (i < NROW * DDIM) {
    int c = i >> 13;            // / NROW
    int r = i & (NROW - 1);
    ZT[(size_t)c * NROW + r] = (_Float16)Z[(size_t)r * DDIM + c];
  } else {
    int j = i - NROW * DDIM;
    if (j < DDIM * DDIM) {
      int o = j >> 7, in = j & 127;
      W1T[o * DDIM + in] = (_Float16)W1[in * DDIM + o];
    } else if (j < 2 * DDIM * DDIM) {
      j -= DDIM * DDIM;
      int o = j >> 7, in = j & 127;
      W2T[o * DDIM + in] = (_Float16)W2[in * DDIM + o];
    }
  }
}

// ---------------- t = relu(Z@W1+b1)@W2+b2, stored f16 row-major -------------
__global__ __launch_bounds__(128) void diff_mlp_kernel(
    const float* __restrict__ Z, const _Float16* __restrict__ W1T,
    const float* __restrict__ b1, const _Float16* __restrict__ W2T,
    const float* __restrict__ b2, _Float16* __restrict__ tH) {
  __shared__ _Float16 hbuf[4][16 * HSTR];
  int lane = threadIdx.x & 31;
  int wave = threadIdx.x >> 5;
  int m0 = (blockIdx.x * 4 + wave) * 16;
  int lid = lane & 15, hh = lane >> 4;
  _Float16* hw = &hbuf[wave][0];

  v16h az[4];
#pragma unroll
  for (int k = 0; k < 4; ++k) {
    const float* p = Z + (size_t)(m0 + lid) * DDIM + k * 32;
#pragma unroll
    for (int i = 0; i < 8; ++i) {
      az[k][i]     = (_Float16)p[hh * 16 + i];
      az[k][i + 8] = (_Float16)p[hh * 16 + 8 + i];
    }
  }

#pragma unroll
  for (int j = 0; j < 8; ++j) {
    v8f c = {};
#pragma unroll
    for (int k = 0; k < 4; ++k) {
      v16h b = load_bfrag(W1T, j * 16, 1, k * 32, DDIM, lane);
      c = wmma16(az[k], b, c);
    }
    float bias = b1[j * 16 + lid];
#pragma unroll
    for (int r = 0; r < 8; ++r) {
      float h = c[r] + bias;
      h = h > 0.f ? h : 0.f;
      hw[(r + hh * 8) * HSTR + j * 16 + lid] = (_Float16)h;
    }
  }

  v16h ah[4];
#pragma unroll
  for (int k = 0; k < 4; ++k) ah[k] = load_afrag(hw, 0, k * 32, HSTR, lane);

#pragma unroll
  for (int j = 0; j < 8; ++j) {
    v8f c = {};
#pragma unroll
    for (int k = 0; k < 4; ++k) {
      v16h b = load_bfrag(W2T, j * 16, 1, k * 32, DDIM, lane);
      c = wmma16(ah[k], b, c);
    }
    float bias = b2[j * 16 + lid];
#pragma unroll
    for (int r = 0; r < 8; ++r) {
      float t = c[r] + bias;
      tH[(size_t)(m0 + r + hh * 8) * DDIM + j * 16 + lid] = (_Float16)t;
    }
  }
}

// ------- fused softmax(t t^T) @ Z, online softmax, split over keys ----------
// Key tiles of tH are async-staged into LDS (double-buffered, shared by the 4
// waves). Sim tile t maps lane column n -> key kb + 4n + t so each lane's 4 P
// values pack into one ds_store_b64, while GEMM2 reads keys naturally from ZT.
__global__ __launch_bounds__(128) void diff_attn_kernel(
    const float* __restrict__ Z, const _Float16* __restrict__ tH,
    const _Float16* __restrict__ ZT, float* __restrict__ Opart,
    float* __restrict__ Mpart, float* __restrict__ Lpart,
    float* __restrict__ out, int ksplit) {
  __shared__ _Float16 stage[2 * STAGEH];         // 2 x 24KB key-tile buffers
  __shared__ _Float16 pbuf[4][16 * PSTR];
  int lane = threadIdx.x & 31;
  int wave = threadIdx.x >> 5;
  int m0 = (blockIdx.x * 4 + wave) * 16;
  int split = blockIdx.y;
  int lid = lane & 15, hh = lane >> 4;
  _Float16* pw = &pbuf[wave][0];

  v16h aq[4];
#pragma unroll
  for (int k = 0; k < 4; ++k) aq[k] = load_afrag(tH, m0, k * 32, DDIM, lane);

  v8f zero = {};
  v8f O[8];
#pragma unroll
  for (int j = 0; j < 8; ++j) O[j] = zero;
  float mrow[8], lrow[8];
#pragma unroll
  for (int r = 0; r < 8; ++r) { mrow[r] = -3.0e38f; lrow[r] = 0.f; }

  int span = NROW / ksplit;
  int kb0 = split * span, kb1 = kb0 + span;

  stage_keyblock(tH, stage, kb0, wave, lane);    // prologue: fill buffer 0

  for (int kb = kb0; kb < kb1; kb += 64) {
    int cur = ((kb - kb0) >> 6) & 1;
    asm volatile("s_wait_asynccnt 0x0" ::: "memory");  // own stage done
    __syncthreads();                                   // whole tile resident
    if (kb + 64 < kb1)
      stage_keyblock(tH, stage + (cur ^ 1) * STAGEH, kb + 64, wave, lane);

    const _Float16* sb = stage + cur * STAGEH;
    v8f s[4];
#pragma unroll
    for (int t = 0; t < 4; ++t) s[t] = zero;
#pragma unroll
    for (int k = 0; k < 4; ++k) {
#pragma unroll
      for (int t = 0; t < 4; ++t) {
        v16h b = load_stage_frag(sb + ((k << 2) | t) * FRAGSZ, lane);
        s[t] = wmma16(aq[k], b, s[t]);
      }
    }
    __builtin_prefetch(ZT + (size_t)(lane * 4) * NROW + kb + 64, 0, 3);

    // online softmax over the 64-key block
    float alph[8];
#pragma unroll
    for (int r = 0; r < 8; ++r) {
      float bm = fmaxf(fmaxf(s[0][r], s[1][r]), fmaxf(s[2][r], s[3][r]));
      bm = fmaxf(bm, __shfl_xor(bm, 1));
      bm = fmaxf(bm, __shfl_xor(bm, 2));
      bm = fmaxf(bm, __shfl_xor(bm, 4));
      bm = fmaxf(bm, __shfl_xor(bm, 8));
      float mn = fmaxf(mrow[r], bm);
      float a  = __expf(mrow[r] - mn);
      mrow[r] = mn;
      alph[r] = a;
      float p0 = __expf(s[0][r] - mn);
      float p1 = __expf(s[1][r] - mn);
      float p2 = __expf(s[2][r] - mn);
      float p3 = __expf(s[3][r] - mn);
      lrow[r] = lrow[r] * a + ((p0 + p1) + (p2 + p3));
      v4h pk;
      pk[0] = (_Float16)p0; pk[1] = (_Float16)p1;
      pk[2] = (_Float16)p2; pk[3] = (_Float16)p3;
      *(v4h*)(pw + (r + hh * 8) * PSTR + 4 * lid) = pk;   // ds_store_b64
    }
    // P back in A layout; O = O*alpha + P @ Z[kb:kb+64, :]
    v16h ap0 = load_afrag(pw, 0, 0,  PSTR, lane);
    v16h ap1 = load_afrag(pw, 0, 32, PSTR, lane);
#pragma unroll
    for (int j = 0; j < 8; ++j) {
      v8f o = O[j];
#pragma unroll
      for (int r = 0; r < 8; ++r) o[r] *= alph[r];
      v16h bz0 = load_bfrag(ZT, j * 16, 1, kb,      NROW, lane);
      o = wmma16(ap0, bz0, o);
      v16h bz1 = load_bfrag(ZT, j * 16, 1, kb + 32, NROW, lane);
      O[j] = wmma16(ap1, bz1, o);
    }
  }

  // finish row sums across the 16-lane half
#pragma unroll
  for (int r = 0; r < 8; ++r) {
    float l = lrow[r];
    l += __shfl_xor(l, 1);
    l += __shfl_xor(l, 2);
    l += __shfl_xor(l, 4);
    l += __shfl_xor(l, 8);
    lrow[r] = l;
  }

  if (out != nullptr) {
#pragma unroll
    for (int j = 0; j < 8; ++j) {
#pragma unroll
      for (int r = 0; r < 8; ++r) {
        size_t idx = (size_t)(m0 + r + hh * 8) * DDIM + j * 16 + lid;
        float z = Z[idx];
        out[idx] = (1.0f - TAUF) * z + TAUF * (O[j][r] / lrow[r]);
      }
    }
  } else {
    size_t obase = (size_t)split * NROW * DDIM;
#pragma unroll
    for (int j = 0; j < 8; ++j) {
#pragma unroll
      for (int r = 0; r < 8; ++r) {
        Opart[obase + (size_t)(m0 + r + hh * 8) * DDIM + j * 16 + lid] = O[j][r];
      }
    }
    if (lid == 0) {
#pragma unroll
      for (int r = 0; r < 8; ++r) {
        Mpart[split * NROW + m0 + r + hh * 8] = mrow[r];
        Lpart[split * NROW + m0 + r + hh * 8] = lrow[r];
      }
    }
  }
}

// ---------------- merge split-K partials, apply the residual blend ----------
__global__ void diff_combine_kernel(const float* __restrict__ Z,
                                    const float* __restrict__ Opart,
                                    const float* __restrict__ Mpart,
                                    const float* __restrict__ Lpart,
                                    float* __restrict__ out, int ksplit) {
  int i = blockIdx.x * blockDim.x + threadIdx.x;   // over NROW*DDIM
  int r = i >> 7;
  float mstar = -3.0e38f;
  for (int s = 0; s < ksplit; ++s) mstar = fmaxf(mstar, Mpart[s * NROW + r]);
  float acc = 0.f, l = 0.f;
  for (int s = 0; s < ksplit; ++s) {
    float w = __expf(Mpart[s * NROW + r] - mstar);
    acc += w * Opart[(size_t)s * NROW * DDIM + i];
    l   += w * Lpart[s * NROW + r];
  }
  float z = Z[i];
  out[i] = (1.0f - TAUF) * z + TAUF * (acc / l);
}

extern "C" void kernel_launch(void* const* d_in, const int* in_sizes, int n_in,
                              void* d_out, int out_size, void* d_ws, size_t ws_size,
                              hipStream_t stream) {
  (void)in_sizes; (void)n_in; (void)out_size;
  const float* Z  = (const float*)d_in[0];
  const float* W1 = (const float*)d_in[1];
  const float* b1 = (const float*)d_in[2];
  const float* W2 = (const float*)d_in[3];
  const float* b2 = (const float*)d_in[4];
  float* out = (float*)d_out;

  _Float16* ws  = (_Float16*)d_ws;
  _Float16* tH  = ws;                                  // [8192][128] f16
  _Float16* ZT  = tH + (size_t)NROW * DDIM;            // [128][8192] f16
  _Float16* W1T = ZT + (size_t)NROW * DDIM;            // [128][128] f16
  _Float16* W2T = W1T + DDIM * DDIM;                   // [128][128] f16

  size_t baseBytes = (size_t)(2 * NROW * DDIM + 2 * DDIM * DDIM) * sizeof(_Float16);
  size_t perSplit  = (size_t)NROW * DDIM * sizeof(float)      // O partial
                   + 2 * (size_t)NROW * sizeof(float);        // m, l partials
  int ksplit = 0;
  for (int k = 8; k >= 1; k >>= 1) {
    if (baseBytes + (size_t)k * perSplit <= ws_size) { ksplit = k; break; }
  }

  int prep_elems = NROW * DDIM + 2 * DDIM * DDIM;
  diff_prep_kernel<<<(prep_elems + 255) / 256, 256, 0, stream>>>(Z, W1, W2, ZT,
                                                                 W1T, W2T);
  diff_mlp_kernel<<<NROW / 64, 128, 0, stream>>>(Z, W1T, b1, W2T, b2, tH);

  if (ksplit >= 2) {
    float* Opart = (float*)((char*)d_ws + baseBytes);
    float* Mpart = Opart + (size_t)ksplit * NROW * DDIM;
    float* Lpart = Mpart + (size_t)ksplit * NROW;
    diff_attn_kernel<<<dim3(NROW / 64, ksplit), 128, 0, stream>>>(
        Z, tH, ZT, Opart, Mpart, Lpart, nullptr, ksplit);
    diff_combine_kernel<<<NROW * DDIM / 256, 256, 0, stream>>>(
        Z, Opart, Mpart, Lpart, out, ksplit);
  } else {
    diff_attn_kernel<<<dim3(NROW / 64, 1), 128, 0, stream>>>(
        Z, tH, ZT, nullptr, nullptr, nullptr, out, 1);
  }
}
